// PaiNNInteraction_62663572848821
// MI455X (gfx1250) — matrix operand order repaired and checked by reading, not compiled, not run on hardware
//
#include <hip/hip_runtime.h>

#define NN   10000
#define NE   320000
#define HID  128
#define H3   384
#define NRBF 20

typedef __attribute__((ext_vector_type(16))) _Float16 v16h;
typedef __attribute__((ext_vector_type(8)))  _Float16 v8h;
typedef __attribute__((ext_vector_type(8)))  float    v8f;

static __device__ __forceinline__ v8f wmma_f16(v16h a, v16h b, v8f c) {
    // D(f32 16x16) = A(f16 16x32) * B(f16 32x16) + C
    return __builtin_amdgcn_wmma_f32_16x16x32_f16(
        /*neg_a=*/false, a, /*neg_b=*/false, b,
        /*c_mod=*/(short)0, c, /*reuse_a=*/false, /*reuse_b=*/false);
}

// A-fragment (16x32 f16): lane holds row m=lane%16; elements 0..15 are
// K = half*16 .. half*16+15.
// rowk points at (row base + kt*32); both chunks are 16B-aligned.
static __device__ __forceinline__ v16h load_afrag(const _Float16* rowk, int half) {
    union { v16h v; v8h h[2]; } u;
    u.h[0] = *(const v8h*)(rowk + half * 16);
    u.h[1] = *(const v8h*)(rowk + half * 16 + 8);
    return u.v;
}

static __device__ __forceinline__ float silu(float v) {
    return v / (1.0f + __expf(-v));
}

static __device__ __forceinline__ void atomic_add_f32(float* p, float v) {
    __hip_atomic_fetch_add(p, v, __ATOMIC_RELAXED, __HIP_MEMORY_SCOPE_AGENT);
}

// ---------------- prep kernels ----------------

__global__ void k_cvt_f16(const float* __restrict__ s, _Float16* __restrict__ d, int n) {
    int i = blockIdx.x * blockDim.x + threadIdx.x;
    if (i < n) d[i] = (_Float16)s[i];
}

// W is [K][N] row-major f32 -> WT is [N][K] f16 (N-major so B-fragments are contiguous)
__global__ void k_transpose_f16(const float* __restrict__ W, _Float16* __restrict__ WT,
                                int K, int N) {
    int i = blockIdx.x * blockDim.x + threadIdx.x;
    if (i < K * N) {
        int n = i / K, k = i % K;
        WT[i] = (_Float16)W[(size_t)k * N + n];
    }
}

// Wf1 [20][128] -> WT [128][32] with K padded 20->32 by zeros
__global__ void k_prep_wf1(const float* __restrict__ Wf1, _Float16* __restrict__ WT) {
    int i = blockIdx.x * blockDim.x + threadIdx.x;
    if (i < HID * 32) {
        int n = i >> 5, k = i & 31;
        WT[i] = (k < NRBF) ? (_Float16)Wf1[(size_t)k * HID + n] : (_Float16)0.f;
    }
}

// out = concat(q [NN*HID], mu [NN*3*HID]); messages are atomically added later
__global__ void k_init_out(const float* __restrict__ q, const float* __restrict__ mu,
                           float* __restrict__ out) {
    int i = blockIdx.x * blockDim.x + threadIdx.x;
    const int nq = NN * HID;
    if (i < nq) out[i] = q[i];
    else if (i < nq + NN * H3) out[i] = mu[i - nq];
}

// ---------------- node MLP: x = silu(q@W1+b1)@W2+b2  (f32 out) ----------------

__global__ __launch_bounds__(128)
void k_node_mlp(const _Float16* __restrict__ qh, const _Float16* __restrict__ W1T,
                const _Float16* __restrict__ W2T, const float* __restrict__ b1,
                const float* __restrict__ b2, float* __restrict__ x) {
    __shared__ _Float16 h_lds[16 * 400];   // [16 nodes][384 cols], stride 400 (16B aligned rows)

    const int tid  = threadIdx.x;
    const int wave = tid >> 5, lane = tid & 31;
    const int half = lane >> 4, n16 = lane & 15;
    const int nodeBase = blockIdx.x * 16;

    const v8f vzero = {0.f,0.f,0.f,0.f,0.f,0.f,0.f,0.f};

    // ---- layer 1: [16,128] @ [128,384] ----
    v8f acc[6];
#pragma unroll
    for (int t = 0; t < 6; ++t) acc[t] = vzero;

#pragma unroll
    for (int kt = 0; kt < 4; ++kt) {
        v16h a = load_afrag(qh + (size_t)(nodeBase + n16) * HID + kt * 32, half);
#pragma unroll
        for (int t = 0; t < 6; ++t) {
            int nt = wave + 4 * t;
            v16h b = *(const v16h*)(W1T + (size_t)(nt * 16 + n16) * HID + kt * 32 + half * 16);
            acc[t] = wmma_f16(a, b, acc[t]);
        }
    }
#pragma unroll
    for (int t = 0; t < 6; ++t) {
        int col = (wave + 4 * t) * 16 + n16;
        float bias = b1[col];
#pragma unroll
        for (int r = 0; r < 8; ++r) {
            float v = silu(acc[t][r] + bias);
            h_lds[(r + 8 * half) * 400 + col] = (_Float16)v;
        }
    }
    __syncthreads();

    // ---- layer 2: [16,384] @ [384,384] ----
    v8f acc2[6];
#pragma unroll
    for (int t = 0; t < 6; ++t) acc2[t] = vzero;

#pragma unroll
    for (int kt = 0; kt < 12; ++kt) {
        v16h a = load_afrag(&h_lds[n16 * 400 + kt * 32], half);
#pragma unroll
        for (int t = 0; t < 6; ++t) {
            int nt = wave + 4 * t;
            v16h b = *(const v16h*)(W2T + (size_t)(nt * 16 + n16) * H3 + kt * 32 + half * 16);
            acc2[t] = wmma_f16(a, b, acc2[t]);
        }
    }
#pragma unroll
    for (int t = 0; t < 6; ++t) {
        int col = (wave + 4 * t) * 16 + n16;
        float bias = b2[col];
#pragma unroll
        for (int r = 0; r < 8; ++r) {
            int mrow = nodeBase + r + 8 * half;
            x[(size_t)mrow * H3 + col] = acc2[t][r] + bias;
        }
    }
}

// ---------------- edge kernel: filter MLP + gather/scatter messages ----------------

__global__ __launch_bounds__(128)
void k_edge(const int* __restrict__ ei, const float* __restrict__ rbf,
            const float* __restrict__ uv, const float* __restrict__ cut,
            const float* __restrict__ bf1, const float* __restrict__ bf2,
            const _Float16* __restrict__ Wf1T, const _Float16* __restrict__ Wf2T,
            const float* __restrict__ x, const float* __restrict__ mu,
            float* __restrict__ out_q, float* __restrict__ out_mu) {
    __shared__ _Float16 rbf_h[16 * 32];    // A tile for GEMM1 (K padded to 32)
    __shared__ _Float16 hf[16 * 144];      // silu(layer1) f16, stride 144 (16B aligned)
    __shared__ float    filt[16 * 384];    // filters f32
    __shared__ int      src_s[16], tgt_s[16];
    __shared__ float    cut_s[16], uv_s[48];

    const int tid  = threadIdx.x;
    const int wave = tid >> 5, lane = tid & 31;
    const int half = lane >> 4, n16 = lane & 15;
    const int eBase = blockIdx.x * 16;

    if (tid < 16) {
        tgt_s[tid] = ei[eBase + tid];          // edge_index[0] = target
        src_s[tid] = ei[NE + eBase + tid];     // edge_index[1] = source
        cut_s[tid] = cut[eBase + tid];
    }
    if (tid < 48) uv_s[tid] = uv[(size_t)eBase * 3 + tid];
    for (int i = tid; i < 16 * 32; i += 128) {
        int e = i >> 5, k = i & 31;
        rbf_h[i] = (k < NRBF) ? (_Float16)rbf[(size_t)(eBase + e) * NRBF + k] : (_Float16)0.f;
    }
    __syncthreads();

    const v8f vzero = {0.f,0.f,0.f,0.f,0.f,0.f,0.f,0.f};

    // ---- GEMM1: [16,32] @ [32,128], single K step; 8 N-tiles over 4 waves ----
    {
        v8f acc[2] = {vzero, vzero};
        v16h a = load_afrag(&rbf_h[n16 * 32], half);
#pragma unroll
        for (int j = 0; j < 2; ++j) {
            int nt = wave * 2 + j;
            v16h b = *(const v16h*)(Wf1T + (size_t)(nt * 16 + n16) * 32 + half * 16);
            acc[j] = wmma_f16(a, b, acc[j]);
        }
#pragma unroll
        for (int j = 0; j < 2; ++j) {
            int col = (wave * 2 + j) * 16 + n16;
            float bias = bf1[col];
#pragma unroll
            for (int r = 0; r < 8; ++r) {
                float v = silu(acc[j][r] + bias);
                hf[(r + 8 * half) * 144 + col] = (_Float16)v;
            }
        }
    }
    __syncthreads();

    // ---- GEMM2: [16,128] @ [128,384]; 24 N-tiles over 4 waves ----
    {
        v8f acc[6];
#pragma unroll
        for (int t = 0; t < 6; ++t) acc[t] = vzero;
#pragma unroll
        for (int kt = 0; kt < 4; ++kt) {
            v16h a = load_afrag(&hf[n16 * 144 + kt * 32], half);
#pragma unroll
            for (int t = 0; t < 6; ++t) {
                int nt = wave + 4 * t;
                v16h b = *(const v16h*)(Wf2T + (size_t)(nt * 16 + n16) * HID + kt * 32 + half * 16);
                acc[t] = wmma_f16(a, b, acc[t]);
            }
        }
#pragma unroll
        for (int t = 0; t < 6; ++t) {
            int col = (wave + 4 * t) * 16 + n16;
            float bias = bf2[col];
#pragma unroll
            for (int r = 0; r < 8; ++r) {
                int m = r + 8 * half;
                filt[m * 384 + col] = (acc[t][r] + bias) * cut_s[m];
            }
        }
    }
    __syncthreads();

    // ---- messages: thread = channel h (0..127), fully coalesced gathers ----
    const int h = tid;
    for (int e = 0; e < 16; ++e) {
        int s = src_s[e], g = tgt_s[e];
        const float* xs  = x  + (size_t)s * H3;
        const float* mus = mu + (size_t)s * H3;
        float fq = filt[e * 384 + h];
        float fr = filt[e * 384 + 128 + h];
        float fm = filt[e * 384 + 256 + h];
        float rr = xs[128 + h] * fr;   // x_r * filter_r
        float mm = xs[256 + h] * fm;   // x_mu * filter_mu

        atomic_add_f32(out_q + (size_t)g * HID + h, xs[h] * fq);
        float* og = out_mu + (size_t)g * H3;
        atomic_add_f32(og + h,       uv_s[e * 3 + 0] * rr + mus[h]       * mm);
        atomic_add_f32(og + 128 + h, uv_s[e * 3 + 1] * rr + mus[128 + h] * mm);
        atomic_add_f32(og + 256 + h, uv_s[e * 3 + 2] * rr + mus[256 + h] * mm);
    }
}

// ---------------- launcher ----------------

extern "C" void kernel_launch(void* const* d_in, const int* in_sizes, int n_in,
                              void* d_out, int out_size, void* d_ws, size_t ws_size,
                              hipStream_t stream) {
    (void)in_sizes; (void)n_in; (void)out_size; (void)ws_size;

    const float* q   = (const float*)d_in[0];
    const float* mu  = (const float*)d_in[1];
    const int*   ei  = (const int*)d_in[2];
    const float* rbf = (const float*)d_in[3];
    const float* uv  = (const float*)d_in[4];
    const float* cut = (const float*)d_in[5];
    const float* W1  = (const float*)d_in[6];
    const float* b1  = (const float*)d_in[7];
    const float* W2  = (const float*)d_in[8];
    const float* b2  = (const float*)d_in[9];
    const float* Wf1 = (const float*)d_in[10];
    const float* bf1 = (const float*)d_in[11];
    const float* Wf2 = (const float*)d_in[12];
    const float* bf2 = (const float*)d_in[13];

    char* ws = (char*)d_ws;
    _Float16* qh   = (_Float16*)(ws + 0);          // 10000*128*2   = 2,560,000
    _Float16* W1T  = (_Float16*)(ws + 2560000);    // 384*128*2     =    98,304
    _Float16* W2T  = (_Float16*)(ws + 2658304);    // 384*384*2     =   294,912
    _Float16* Wf1T = (_Float16*)(ws + 2953216);    // 128*32*2      =     8,192
    _Float16* Wf2T = (_Float16*)(ws + 2961408);    // 384*128*2     =    98,304
    float*    x    = (float*)   (ws + 3059712);    // 10000*384*4   = 15,360,000

    float* out_q  = (float*)d_out;
    float* out_mu = out_q + (size_t)NN * HID;

    k_cvt_f16<<<(NN * HID + 255) / 256, 256, 0, stream>>>(q, qh, NN * HID);
    k_transpose_f16<<<(HID * H3 + 255) / 256, 256, 0, stream>>>(W1, W1T, HID, H3);
    k_transpose_f16<<<(H3 * H3 + 255) / 256, 256, 0, stream>>>(W2, W2T, H3, H3);
    k_transpose_f16<<<(HID * H3 + 255) / 256, 256, 0, stream>>>(Wf2, Wf2T, HID, H3);
    k_prep_wf1<<<(HID * 32 + 255) / 256, 256, 0, stream>>>(Wf1, Wf1T);
    k_init_out<<<(NN * (HID + H3) + 255) / 256, 256, 0, stream>>>(q, mu, (float*)d_out);

    k_node_mlp<<<NN / 16, 128, 0, stream>>>(qh, W1T, W2T, b1, b2, x);
    k_edge<<<NE / 16, 128, 0, stream>>>(ei, rbf, uv, cut, bf1, bf2, Wf1T, Wf2T,
                                        x, mu, out_q, out_mu);
}
